// DGCNN_cls_83305185673220
// MI455X (gfx1250) — matrix operand order repaired and checked by reading, not compiled, not run on hardware
//
#include <hip/hip_runtime.h>
#include <math.h>

#define LSLOPE 0.2f
#define BN_EPS 1e-5f

typedef __attribute__((ext_vector_type(16))) __bf16 v16bf;
typedef __attribute__((ext_vector_type(8)))  float  v8f;

// ---- helpers -------------------------------------------------------------

// native f32 -> bf16 (RNE); lowers to gfx1250 bf16 cvt ops
__device__ inline __bf16 bfbits(float f) { return (__bf16)f; }

// order-preserving float <-> uint mapping for atomicMax-based max-reductions
__device__ inline unsigned fenc(float f) {
  unsigned u = __float_as_uint(f);
  return (u & 0x80000000u) ? ~u : (u | 0x80000000u);
}
__device__ inline float fdec(unsigned u) {
  return __uint_as_float((u & 0x80000000u) ? (u ^ 0x80000000u) : ~u);
}

// Fragment layouts per CDNA5 ISA 7.12.2 (wave32), 16x16x32 bf16:
//   A: lane L holds M=L%16; elem e -> K = e + 8*(L/16) + (e>=8 ? 8 : 0)
//      => two contiguous 8-float runs at rowp+8h and rowp+16+8h
//   B: lane L holds N=L%16; elem e -> K = e + 8*(L/16) + (e>=8 ? 8 : 0)
//      => two contiguous 8-float runs at rowp+8h and rowp+16+8h
//   C/D: lane L holds N=L%16, rows M = j + 8*(L/16), j=0..7
__device__ inline v16bf frag_a(const float* __restrict__ rowp, int half) {
  const float* p = rowp + 8 * half;
  float4 a0 = *(const float4*)(p);
  float4 a1 = *(const float4*)(p + 4);
  float4 b0 = *(const float4*)(p + 16);
  float4 b1 = *(const float4*)(p + 20);
  v16bf A;
  A[0]  = bfbits(a0.x); A[1]  = bfbits(a0.y); A[2]  = bfbits(a0.z); A[3]  = bfbits(a0.w);
  A[4]  = bfbits(a1.x); A[5]  = bfbits(a1.y); A[6]  = bfbits(a1.z); A[7]  = bfbits(a1.w);
  A[8]  = bfbits(b0.x); A[9]  = bfbits(b0.y); A[10] = bfbits(b0.z); A[11] = bfbits(b0.w);
  A[12] = bfbits(b1.x); A[13] = bfbits(b1.y); A[14] = bfbits(b1.z); A[15] = bfbits(b1.w);
  return A;
}
__device__ inline v16bf frag_b16(const float* __restrict__ rowp, int half) {
  const float* p = rowp + 8 * half;
  float4 c0 = *(const float4*)(p);
  float4 c1 = *(const float4*)(p + 4);
  float4 c2 = *(const float4*)(p + 16);
  float4 c3 = *(const float4*)(p + 20);
  v16bf Bv;
  Bv[0]  = bfbits(c0.x); Bv[1]  = bfbits(c0.y); Bv[2]  = bfbits(c0.z); Bv[3]  = bfbits(c0.w);
  Bv[4]  = bfbits(c1.x); Bv[5]  = bfbits(c1.y); Bv[6]  = bfbits(c1.z); Bv[7]  = bfbits(c1.w);
  Bv[8]  = bfbits(c2.x); Bv[9]  = bfbits(c2.y); Bv[10] = bfbits(c2.z); Bv[11] = bfbits(c2.w);
  Bv[12] = bfbits(c3.x); Bv[13] = bfbits(c3.y); Bv[14] = bfbits(c3.z); Bv[15] = bfbits(c3.w);
  return Bv;
}

// ---- elementwise / utility kernels --------------------------------------

__global__ void zero_f_kernel(float* p, size_t n) {
  size_t i = (size_t)blockIdx.x * blockDim.x + threadIdx.x;
  if (i < n) p[i] = 0.f;
}
__global__ void zero_u_kernel(unsigned* p, size_t n) {
  size_t i = (size_t)blockIdx.x * blockDim.x + threadIdx.x;
  if (i < n) p[i] = 0u;
}
__global__ void zero2_kernel(unsigned* p, int B, int stride, int off, int O) {
  int i = blockIdx.x * blockDim.x + threadIdx.x;
  if (i >= B * O) return;
  int b = i / O, o = i - b * O;
  p[(size_t)b * stride + off + o] = 0u;
}
__global__ void decode_kernel(float* p, size_t n) {
  size_t i = (size_t)blockIdx.x * blockDim.x + threadIdx.x;
  if (i >= n) return;
  unsigned u = ((unsigned*)p)[i];
  p[i] = fdec(u);
}
__global__ void decode2_kernel(float* p, int B, int stride, int off, int O) {
  int i = blockIdx.x * blockDim.x + threadIdx.x;
  if (i >= B * O) return;
  int b = i / O, o = i - b * O;
  size_t j = (size_t)b * stride + off + o;
  unsigned u = ((unsigned*)p)[j];
  p[j] = fdec(u);
}

__global__ void norms_kernel(const float* __restrict__ X, float* __restrict__ xx,
                             int B, int C, int N) {
  int i = blockIdx.x * blockDim.x + threadIdx.x;
  if (i >= B * N) return;
  int b = i / N, n = i - b * N;
  const float* p = X + (size_t)b * C * N + n;
  float s = 0.f;
  for (int c = 0; c < C; ++c) { float v = p[(size_t)c * N]; s += v * v; }
  xx[i] = s;
}

__global__ void copych_kernel(float* __restrict__ dst, const float* __restrict__ src,
                              int B, int Cd, int coff, int Cs, int N) {
  int i = blockIdx.x * blockDim.x + threadIdx.x;
  if (i >= B * Cs * N) return;
  int b = i / (Cs * N);
  int r = i - b * Cs * N;
  int c = r / N, n = r - c * N;
  dst[(size_t)b * Cd * N + (size_t)(coff + c) * N + n] = src[i];
}

// ---- fused Gram (WMMA) + top-k kNN --------------------------------------
// One workgroup per (batch, 16-row stripe); 16 waves. For each 256-col chunk:
// cooperatively stage X-slab into LDS (coalesced b128, zero-padded K), each
// wave computes one 16x16 dot tile per K-step via WMMA with b128 LDS frags,
// then 16 threads do stable top-k insertion over the distance chunk.
#define MCH 256
__global__ __launch_bounds__(512) void knn_kernel(const float* __restrict__ X,
                                                  const float* __restrict__ xx,
                                                  int* __restrict__ idx,
                                                  int C, int N, int ksel) {
  __shared__ float xs[MCH][36];        // B slab: [m][k], padded
  __shared__ float as[16][36];         // A slab: [r][k], padded
  __shared__ float dist[16][MCH + 1];
  const int tid = threadIdx.x;
  const int wid = tid >> 5, lane = tid & 31, lm = lane & 15, half = lane >> 4;
  const int ntiles = N >> 4;
  const int b  = blockIdx.x / ntiles;
  const int n0 = (blockIdx.x % ntiles) << 4;
  const float* Xb  = X + (size_t)b * C * N;
  const float* xxb = xx + (size_t)b * N;

  float tv[20]; int ti[20];
  if (tid < 16)
    for (int j = 0; j < 20; ++j) { tv[j] = -3.0e38f; ti[j] = 0; }

  const int kc = (C + 31) >> 5;
  for (int m0c = 0; m0c < N; m0c += MCH) {
    const int mcnt   = min(MCH, N - m0c);
    const int mtiles = mcnt >> 4;
    v8f acc = {0.f, 0.f, 0.f, 0.f, 0.f, 0.f, 0.f, 0.f};
    for (int q = 0; q < kc; ++q) {
      const int kb = q << 5;
      __syncthreads();                         // protect slab reuse
      // stage B slab: 32 k-rows x MCH cols, float4 along m, zero-padded
      for (int i = tid; i < 32 * (MCH / 4); i += 512) {
        int kk = i >> 6, m4 = (i & 63) << 2;
        int c = kb + kk;
        float4 v = {0.f, 0.f, 0.f, 0.f};
        if (c < C && m4 < mcnt) {
          v = *(const float4*)(Xb + (size_t)c * N + m0c + m4);
          __builtin_prefetch(Xb + (size_t)c * N + m0c + m4 + MCH, 0, 1);
        }
        xs[m4 + 0][kk] = v.x; xs[m4 + 1][kk] = v.y;
        xs[m4 + 2][kk] = v.z; xs[m4 + 3][kk] = v.w;
      }
      // stage A slab: 32 k-rows x 16 rows (one element per thread)
      {
        int kk = tid >> 4, r = tid & 15;
        int c = kb + kk;
        as[r][kk] = (c < C) ? Xb[(size_t)c * N + n0 + r] : 0.f;
      }
      __syncthreads();
      if (wid < mtiles) {
        v16bf A  = frag_a(&as[lm][0], half);
        v16bf Bv = frag_b16(&xs[wid * 16 + lm][0], half);
        acc = __builtin_amdgcn_wmma_f32_16x16x32_bf16(false, A, false, Bv,
                                                      (short)0, acc, false, false);
      }
    }
    if (wid < mtiles) {
      const int m = m0c + wid * 16 + lm;
#pragma unroll
      for (int j = 0; j < 8; ++j) {
        const int r = j + 8 * half;
        dist[r][wid * 16 + lm] = 2.f * acc[j] - xxb[n0 + r] - xxb[m];
      }
    }
    __syncthreads();
    if (tid < 16) {
      for (int m = 0; m < mcnt; ++m) {
        float d = dist[tid][m];
        if (d > tv[ksel - 1]) {                // strict > keeps jax tie order
          int pos = ksel - 1;
          while (pos > 0 && d > tv[pos - 1]) {
            tv[pos] = tv[pos - 1]; ti[pos] = ti[pos - 1]; --pos;
          }
          tv[pos] = d; ti[pos] = m0c + m;
        }
      }
    }
    __syncthreads();
  }
  if (tid < 16) {
    int* op = idx + ((size_t)b * N + n0 + tid) * ksel;
    for (int j = 0; j < ksel; ++j) op[j] = ti[j];
  }
}

// ---- edge conv: W(O x 2C) @ graph_feature, two passes -------------------
// Block = 4 waves sharing one gathered 16-col feature tile staged in LDS.
// pass 1: per-channel sum/sumsq (BN stats), shuffle-reduced then f32 atomics
// pass 2: BN + LeakyReLU + max-over-k via encoded atomicMax into outu(B,O,N)
__global__ __launch_bounds__(128) void edgeconv_kernel(
    const float* __restrict__ X, const int* __restrict__ idx,
    const float* __restrict__ W,
    float* __restrict__ sum, float* __restrict__ sumsq,
    const float* __restrict__ mean, const float* __restrict__ inva,
    unsigned* __restrict__ outu,
    int C, int N, int ksel, int O, int pass) {
  __shared__ float fs[16][36];                 // gathered tile: [col][k]
  const int tid = threadIdx.x;
  const int wid = tid >> 5, lane = tid & 31, lm = lane & 15, half = lane >> 4;
  const int ctiles  = (N * ksel) >> 4;
  const int ogroups = O >> 6;                  // O / (4 waves * 16)
  int t = blockIdx.x;
  const int col0 = (t % ctiles) << 4; t /= ctiles;
  const int o0   = ((t % ogroups) << 6) + wid * 16;
  const int b    = t / ogroups;
  const float* Xb = X + (size_t)b * C * N;
  const int* idxb = idx + (size_t)b * N * ksel;
  const int C2 = 2 * C;
  const int kc = (C2 + 31) >> 5;
  const bool fastA = (C2 & 31) == 0;
  v8f acc = {0.f, 0.f, 0.f, 0.f, 0.f, 0.f, 0.f, 0.f};
  for (int q = 0; q < kc; ++q) {
    const int kb = q << 5;
    __syncthreads();
    for (int i = tid; i < 512; i += 128) {     // 32 k x 16 cols, gather once
      int col = i & 15, kk = i >> 4;
      int c = kb + kk;
      float v = 0.f;
      if (c < C2) {
        int cg = col0 + col;
        int n = cg / ksel, j = cg - n * ksel;
        if (c < C) {
          int m = idxb[(size_t)n * ksel + j];
          v = Xb[(size_t)c * N + m] - Xb[(size_t)c * N + n];
        } else {
          v = Xb[(size_t)(c - C) * N + n];
        }
      }
      fs[col][kk] = v;
    }
    __syncthreads();
    v16bf A;
    if (fastA) {
      A = frag_a(W + (size_t)(o0 + lm) * C2 + kb, half);
    } else {
#pragma unroll
      for (int e = 0; e < 16; ++e) {
        int c = kb + e + 8 * half + (e >= 8 ? 8 : 0);
        A[e] = bfbits((c < C2) ? W[(size_t)(o0 + lm) * C2 + c] : 0.f);
      }
    }
    v16bf Bv = frag_b16(&fs[lm][0], half);
    acc = __builtin_amdgcn_wmma_f32_16x16x32_bf16(false, A, false, Bv,
                                                  (short)0, acc, false, false);
  }
  if (pass == 1) {
#pragma unroll
    for (int j = 0; j < 8; ++j) {
      float v = acc[j], v2 = v * v;
      for (int m = 8; m >= 1; m >>= 1) {
        v += __shfl_xor(v, m, 32); v2 += __shfl_xor(v2, m, 32);
      }
      if (lm == 0) {
        int o = o0 + j + 8 * half;
        atomicAdd(&sum[o], v);
        atomicAdd(&sumsq[o], v2);
      }
    }
  } else {
    const int n = (col0 + lm) / ksel;
#pragma unroll
    for (int j = 0; j < 8; ++j) {
      int o = o0 + j + 8 * half;
      float h = (acc[j] - mean[o]) * inva[o];
      h = (h >= 0.f) ? h : LSLOPE * h;
      atomicMax(&outu[((size_t)b * O + o) * N + n], fenc(h));
    }
  }
}

__global__ void finalize_stats_kernel(const float* sum, const float* sumsq,
                                      float* mean, float* inva, int O, float cnt) {
  int o = blockIdx.x * blockDim.x + threadIdx.x;
  if (o >= O) return;
  float m = sum[o] / cnt;
  float v = sumsq[o] / cnt - m * m;
  mean[o] = m;
  inva[o] = rsqrtf(v + BN_EPS);
}

// ---- 1x1 conv (WMMA): W(O x C) @ X(B,C,N) -------------------------------
// Block = 8 waves (8 o-tiles) sharing one staged X tile in LDS.
// mode 0: BN stats; mode 1: BN+LReLU, max-over-n -> outu[b*ostride+ooff+o]
// mode 2: +bias, max-over-n -> outu (pool projection vector)
__global__ __launch_bounds__(256) void conv1_kernel(
    const float* __restrict__ X, const float* __restrict__ W,
    const float* __restrict__ bias,
    float* __restrict__ sum, float* __restrict__ sumsq,
    const float* __restrict__ mean, const float* __restrict__ inva,
    unsigned* __restrict__ outu, int ostride, int ooff,
    int C, int N, int O, int mode) {
  __shared__ float xsh[16][36];                // X tile: [n][k]
  const int tid = threadIdx.x;
  const int wid = tid >> 5, lane = tid & 31, lm = lane & 15, half = lane >> 4;
  const int ctiles  = N >> 4;
  const int ogroups = O >> 7;                  // O / (8 waves * 16)
  int t = blockIdx.x;
  const int n0 = (t % ctiles) << 4; t /= ctiles;
  const int o0 = ((t % ogroups) << 7) + wid * 16;
  const int b  = t / ogroups;
  const float* Xb = X + (size_t)b * C * N;
  const int kc = C >> 5;
  v8f acc = {0.f, 0.f, 0.f, 0.f, 0.f, 0.f, 0.f, 0.f};
  for (int q = 0; q < kc; ++q) {
    const int kb = q << 5;
    __syncthreads();
    for (int i = tid; i < 128; i += 256) {     // 32 k x 16 n, float4 along n
      int kk = i >> 2, n4 = (i & 3) << 2;
      float4 v = *(const float4*)(Xb + (size_t)(kb + kk) * N + n0 + n4);
      xsh[n4 + 0][kk] = v.x; xsh[n4 + 1][kk] = v.y;
      xsh[n4 + 2][kk] = v.z; xsh[n4 + 3][kk] = v.w;
    }
    __syncthreads();
    v16bf A  = frag_a(W + (size_t)(o0 + lm) * C + kb, half);
    v16bf Bv = frag_b16(&xsh[lm][0], half);
    acc = __builtin_amdgcn_wmma_f32_16x16x32_bf16(false, A, false, Bv,
                                                  (short)0, acc, false, false);
  }
  if (mode == 0) {
#pragma unroll
    for (int j = 0; j < 8; ++j) {
      float v = acc[j], v2 = v * v;
      for (int m = 8; m >= 1; m >>= 1) {
        v += __shfl_xor(v, m, 32); v2 += __shfl_xor(v2, m, 32);
      }
      if (lm == 0) {
        int o = o0 + j + 8 * half;
        atomicAdd(&sum[o], v);
        atomicAdd(&sumsq[o], v2);
      }
    }
  } else if (mode == 1) {
#pragma unroll
    for (int j = 0; j < 8; ++j) {
      int o = o0 + j + 8 * half;
      float h = (acc[j] - mean[o]) * inva[o];
      h = (h >= 0.f) ? h : LSLOPE * h;
      atomicMax(&outu[(size_t)b * ostride + ooff + o], fenc(h));
    }
  } else {
#pragma unroll
    for (int j = 0; j < 8; ++j) {
      int o = o0 + j + 8 * half;
      float h = acc[j] + bias[o];
      atomicMax(&outu[(size_t)b * ostride + ooff + o], fenc(h));
    }
  }
}

// ---- pooling helpers -----------------------------------------------------

__global__ void score_kernel(const float* __restrict__ feat,
                             const float* __restrict__ vec,
                             float* __restrict__ sc, int B, int C, int N) {
  int i = blockIdx.x * blockDim.x + threadIdx.x;
  if (i >= B * N) return;
  int b = i / N, n = i - b * N;
  const float* f = feat + (size_t)b * C * N + n;
  const float* v = vec + (size_t)b * C;
  float s = 0.f;
  for (int c = 0; c < C; ++c) s += f[(size_t)c * N] * v[c];
  sc[i] = 1.f / (1.f + expf(-s));
}

__global__ void topk_kernel(const float* __restrict__ sc,
                            float* __restrict__ vals, int* __restrict__ ids,
                            int N, int kp) {
  int b = blockIdx.x;
  if (threadIdx.x != 0) return;
  float tv[256]; int ti[256];
  for (int j = 0; j < kp; ++j) { tv[j] = -3.0e38f; ti[j] = 0; }
  const float* s = sc + (size_t)b * N;
  for (int m = 0; m < N; ++m) {
    float d = s[m];
    if (d > tv[kp - 1]) {
      int pos = kp - 1;
      while (pos > 0 && d > tv[pos - 1]) {
        tv[pos] = tv[pos - 1]; ti[pos] = ti[pos - 1]; --pos;
      }
      tv[pos] = d; ti[pos] = m;
    }
  }
  for (int j = 0; j < kp; ++j) {
    vals[(size_t)b * kp + j] = tv[j];
    ids[(size_t)b * kp + j]  = ti[j];
  }
}

__global__ void gather_scale_kernel(float* __restrict__ dst,
                                    const float* __restrict__ src,
                                    const int* __restrict__ ids,
                                    const float* __restrict__ vals,
                                    int B, int C, int Ns, int kp) {
  int i = blockIdx.x * blockDim.x + threadIdx.x;
  if (i >= B * C * kp) return;
  int b = i / (C * kp);
  int r = i - b * C * kp;
  int c = r / kp, j = r - c * kp;
  int m = ids[(size_t)b * kp + j];
  dst[i] = src[(size_t)b * C * Ns + (size_t)c * Ns + m] * vals[(size_t)b * kp + j];
}

// ---- FC head (WMMA, M = 16 batch rows) ----------------------------------

__global__ __launch_bounds__(32) void fc_kernel(
    const float* __restrict__ Xin,    // (16, C)
    const float* __restrict__ W,      // (O, C)
    const float* __restrict__ bias,
    float* __restrict__ out,          // (16, O)
    int C, int O) {
  const int lane = threadIdx.x & 31, lm = lane & 15, half = lane >> 4;
  const int o0 = blockIdx.x << 4;
  const int oc = min(o0 + lm, O - 1);          // clamp, store is guarded
  const int kc = C >> 5;
  v8f acc = {0.f, 0.f, 0.f, 0.f, 0.f, 0.f, 0.f, 0.f};
  for (int q = 0; q < kc; ++q) {
    const int kb = q << 5;
    v16bf A  = frag_a(Xin + (size_t)lm * C + kb, half);
    v16bf Bv = frag_b16(W + (size_t)oc * C + kb, half);
    acc = __builtin_amdgcn_wmma_f32_16x16x32_bf16(false, A, false, Bv,
                                                  (short)0, acc, false, false);
  }
  const int o = o0 + lm;
  if (o < O) {
    float bb = bias ? bias[o] : 0.f;
#pragma unroll
    for (int j = 0; j < 8; ++j) {
      int row = j + 8 * half;
      out[(size_t)row * O + o] = acc[j] + bb;
    }
  }
}

__global__ void colstats_kernel(const float* __restrict__ h, float* mean,
                                float* inva, int B, int O) {
  int o = blockIdx.x * blockDim.x + threadIdx.x;
  if (o >= O) return;
  float s = 0.f, s2 = 0.f;
  for (int b = 0; b < B; ++b) {
    float v = h[(size_t)b * O + o]; s += v; s2 += v * v;
  }
  float m = s / B;
  mean[o] = m;
  inva[o] = rsqrtf(s2 / B - m * m + BN_EPS);
}

__global__ void bnlrelu_kernel(const float* __restrict__ in, float* __restrict__ out,
                               const float* mean, const float* inva, int B, int O) {
  int i = blockIdx.x * blockDim.x + threadIdx.x;
  if (i >= B * O) return;
  int o = i % O;
  float h = (in[i] - mean[o]) * inva[o];
  out[i] = (h >= 0.f) ? h : LSLOPE * h;
}

// ---- host orchestration --------------------------------------------------

extern "C" void kernel_launch(void* const* d_in, const int* in_sizes, int n_in,
                              void* d_out, int out_size, void* d_ws, size_t ws_size,
                              hipStream_t stream) {
  (void)in_sizes; (void)n_in; (void)out_size; (void)ws_size;

  const float* x   = (const float*)d_in[0];
  const float* w1  = (const float*)d_in[1];
  const float* w2  = (const float*)d_in[2];
  const float* w2m = (const float*)d_in[3];
  const float* w3  = (const float*)d_in[4];
  const float* w4  = (const float*)d_in[5];
  const float* w4m = (const float*)d_in[6];
  const float* w5  = (const float*)d_in[7];
  const float* w5m = (const float*)d_in[8];
  const float* wp1 = (const float*)d_in[9];
  const float* bp1 = (const float*)d_in[10];
  const float* wp2 = (const float*)d_in[11];
  const float* bp2 = (const float*)d_in[12];
  const float* wl1 = (const float*)d_in[13];
  const float* wl2 = (const float*)d_in[14];
  const float* bl2 = (const float*)d_in[15];
  const float* wl3 = (const float*)d_in[16];
  const float* bl3 = (const float*)d_in[17];

  float* outp  = (float*)d_out;          // (16,40)
  float* node1 = outp + 16 * 40;         // (16,3,256), stays scaled
  float* node2 = node1 + 16 * 3 * 256;   // (16,3,64)

  const int Bn = 16;

  float* wsf = (float*)d_ws;
  size_t off = 0;
  auto alloc = [&](size_t n) { float* p = wsf + off; off += n; return p; };
  float* xx    = alloc((size_t)16 * 2048);
  int*   idxb  = (int*)alloc((size_t)16 * 2048 * 20);
  float* sumb  = alloc(1024);
  float* sumq  = alloc(1024);          // contiguous after sumb
  float* meanb = alloc(1024);
  float* invab = alloc(1024);
  float* x1    = alloc((size_t)16 * 64 * 2048);
  float* x2    = alloc((size_t)16 * 64 * 2048);
  float* xt1   = alloc((size_t)16 * 128 * 2048);
  float* gbuf  = alloc((size_t)16 * 3072);
  float* scbuf = alloc((size_t)16 * 2048);
  float* vecb  = alloc((size_t)16 * 256);
  float* pvals = alloc((size_t)16 * 256);
  int*   pids  = (int*)alloc((size_t)16 * 256);
  float* nf1   = alloc((size_t)16 * 128 * 256);
  float* x3    = alloc((size_t)16 * 128 * 256);
  float* x4    = alloc((size_t)16 * 128 * 256);
  float* xt2   = alloc((size_t)16 * 256 * 256);
  float* nf2   = alloc((size_t)16 * 256 * 64);
  float* x5    = alloc((size_t)16 * 256 * 64);
  float* h1    = alloc((size_t)16 * 512);
  float* h1a   = alloc((size_t)16 * 512);
  float* h2    = alloc((size_t)16 * 256);
  float* h2a   = alloc((size_t)16 * 256);

  auto gz = [](size_t n) { return (unsigned)((n + 255) / 256); };

  // full edge-conv stage: knn -> stats pass -> bn/lrelu/max pass
  auto edge_stage = [&](const float* Xf, const float* Wm, int C, int N, int ks,
                        int O, float* xout) {
    norms_kernel<<<gz((size_t)Bn * N), 256, 0, stream>>>(Xf, xx, Bn, C, N);
    knn_kernel<<<Bn * (N / 16), 512, 0, stream>>>(Xf, xx, idxb, C, N, ks);
    zero_f_kernel<<<gz(2048), 256, 0, stream>>>(sumb, 2048);
    int blocks = Bn * (O / 64) * ((N * ks) / 16);
    edgeconv_kernel<<<blocks, 128, 0, stream>>>(Xf, idxb, Wm, sumb, sumq,
        nullptr, nullptr, nullptr, C, N, ks, O, 1);
    finalize_stats_kernel<<<gz(O), 256, 0, stream>>>(sumb, sumq, meanb, invab,
        O, (float)Bn * N * ks);
    size_t on = (size_t)Bn * O * N;
    zero_u_kernel<<<gz(on), 256, 0, stream>>>((unsigned*)xout, on);
    edgeconv_kernel<<<blocks, 128, 0, stream>>>(Xf, idxb, Wm, nullptr, nullptr,
        meanb, invab, (unsigned*)xout, C, N, ks, O, 2);
    decode_kernel<<<gz(on), 256, 0, stream>>>(xout, on);
  };

  // fused conv1 + BN + LReLU + max-over-n into gbuf[:, goff:goff+1024]
  auto conv1_bn_max = [&](const float* Xf, const float* Wm, int C, int N, int goff) {
    const int O = 1024;
    zero_f_kernel<<<gz(2048), 256, 0, stream>>>(sumb, 2048);
    int blocks = Bn * (O / 128) * (N / 16);
    conv1_kernel<<<blocks, 256, 0, stream>>>(Xf, Wm, nullptr, sumb, sumq,
        nullptr, nullptr, nullptr, 0, 0, C, N, O, 0);
    finalize_stats_kernel<<<gz(O), 256, 0, stream>>>(sumb, sumq, meanb, invab,
        O, (float)Bn * N);
    zero2_kernel<<<gz((size_t)Bn * O), 256, 0, stream>>>((unsigned*)gbuf, Bn, 3072, goff, O);
    conv1_kernel<<<blocks, 256, 0, stream>>>(Xf, Wm, nullptr, nullptr, nullptr,
        meanb, invab, (unsigned*)gbuf, 3072, goff, C, N, O, 1);
    decode2_kernel<<<gz((size_t)Bn * O), 256, 0, stream>>>(gbuf, Bn, 3072, goff, O);
  };

  // ---- stage 1: x (C=3) -> x1 (16,64,2048)
  edge_stage(x, w1, 3, 2048, 20, 64, x1);
  // ---- stage 2: x1 (C=64) -> x2
  edge_stage(x1, w2, 64, 2048, 20, 64, x2);
  // xt1_ = concat(x1, x2)
  copych_kernel<<<gz((size_t)Bn * 64 * 2048), 256, 0, stream>>>(xt1, x1, Bn, 128, 0, 64, 2048);
  copych_kernel<<<gz((size_t)Bn * 64 * 2048), 256, 0, stream>>>(xt1, x2, Bn, 128, 64, 64, 2048);
  // g[:, 0:1024] = max_n lrelu(bn(w2m @ xt1_))
  conv1_bn_max(xt1, w2m, 128, 2048, 0);

  // ---- pool 1 (kp=256)
  zero_u_kernel<<<gz((size_t)Bn * 128), 256, 0, stream>>>((unsigned*)vecb, (size_t)Bn * 128);
  conv1_kernel<<<Bn * (128 / 128) * (2048 / 16), 256, 0, stream>>>(xt1, wp1, bp1,
      nullptr, nullptr, nullptr, nullptr, (unsigned*)vecb, 128, 0, 128, 2048, 128, 2);
  decode_kernel<<<gz((size_t)Bn * 128), 256, 0, stream>>>(vecb, (size_t)Bn * 128);
  score_kernel<<<gz((size_t)Bn * 2048), 256, 0, stream>>>(xt1, vecb, scbuf, Bn, 128, 2048);
  topk_kernel<<<Bn, 1, 0, stream>>>(scbuf, pvals, pids, 2048, 256);
  gather_scale_kernel<<<gz((size_t)Bn * 3 * 256), 256, 0, stream>>>(node1, x, pids, pvals, Bn, 3, 2048, 256);
  gather_scale_kernel<<<gz((size_t)Bn * 128 * 256), 256, 0, stream>>>(nf1, xt1, pids, pvals, Bn, 128, 2048, 256);

  // ---- stage 3: nf1 (C=128, N=256, k=10) -> x3
  edge_stage(nf1, w3, 128, 256, 10, 128, x3);
  // ---- stage 4: x3 -> x4
  edge_stage(x3, w4, 128, 256, 10, 128, x4);
  copych_kernel<<<gz((size_t)Bn * 128 * 256), 256, 0, stream>>>(xt2, x3, Bn, 256, 0, 128, 256);
  copych_kernel<<<gz((size_t)Bn * 128 * 256), 256, 0, stream>>>(xt2, x4, Bn, 256, 128, 128, 256);
  // g[:, 1024:2048]
  conv1_bn_max(xt2, w4m, 256, 256, 1024);

  // ---- pool 2 (kp=64)
  zero_u_kernel<<<gz((size_t)Bn * 256), 256, 0, stream>>>((unsigned*)vecb, (size_t)Bn * 256);
  conv1_kernel<<<Bn * (256 / 128) * (256 / 16), 256, 0, stream>>>(xt2, wp2, bp2,
      nullptr, nullptr, nullptr, nullptr, (unsigned*)vecb, 256, 0, 256, 256, 256, 2);
  decode_kernel<<<gz((size_t)Bn * 256), 256, 0, stream>>>(vecb, (size_t)Bn * 256);
  score_kernel<<<gz((size_t)Bn * 256), 256, 0, stream>>>(xt2, vecb, scbuf, Bn, 256, 256);
  topk_kernel<<<Bn, 1, 0, stream>>>(scbuf, pvals, pids, 256, 64);
  gather_scale_kernel<<<gz((size_t)Bn * 3 * 64), 256, 0, stream>>>(node2, node1, pids, pvals, Bn, 3, 256, 64);
  gather_scale_kernel<<<gz((size_t)Bn * 256 * 64), 256, 0, stream>>>(nf2, xt2, pids, pvals, Bn, 256, 256, 64);

  // ---- stage 5: nf2 (C=256, N=64, k=5) -> x5
  edge_stage(nf2, w5, 256, 64, 5, 256, x5);
  // g[:, 2048:3072]
  conv1_bn_max(x5, w5m, 256, 64, 2048);

  // ---- FC head
  fc_kernel<<<512 / 16, 32, 0, stream>>>(gbuf, wl1, nullptr, h1, 3072, 512);
  colstats_kernel<<<gz(512), 256, 0, stream>>>(h1, meanb, invab, Bn, 512);
  bnlrelu_kernel<<<gz((size_t)Bn * 512), 256, 0, stream>>>(h1, h1a, meanb, invab, Bn, 512);
  fc_kernel<<<256 / 16, 32, 0, stream>>>(h1a, wl2, bl2, h2, 512, 256);
  colstats_kernel<<<gz(256), 256, 0, stream>>>(h2, meanb, invab, Bn, 256);
  bnlrelu_kernel<<<gz((size_t)Bn * 256), 256, 0, stream>>>(h2, h2a, meanb, invab, Bn, 256);
  fc_kernel<<<3, 32, 0, stream>>>(h2a, wl3, bl3, outp, 256, 40);
}
